// SigLipLoss_2259152797804
// MI455X (gfx1250) — matrix operand order repaired and checked by reading, not compiled, not run on hardware
//
#include <hip/hip_runtime.h>
#include <hip/hip_bf16.h>

typedef __attribute__((ext_vector_type(16))) __bf16 v16bf;
typedef __attribute__((ext_vector_type(8)))  float  v8f;
typedef __attribute__((ext_vector_type(4)))  float  v4f;

#define NTOT 16384
#define DDIM 512
#define BM 256                   // block tile rows (image)
#define BN 128                   // block tile cols (text)
#define BK 32
#define PITCH 48                 // 32 + 16 pad: 16B-aligned b128, minimal bank conflicts
#define A_TILE (BM * PITCH)      // 12288 bf16 per precision
#define B_TILE (BN * PITCH)      // 6144 bf16 per precision

#define AHI 0
#define ALO (A_TILE)
#define BHI (2 * A_TILE)
#define BLO (2 * A_TILE + B_TILE)
#define SM_ELEMS (2 * A_TILE + 2 * B_TILE)   // 36864 bf16 = 72 KB

// A fragment (16x32 bf16, row-major in LDS):
// lane m=lane&15 holds row M=m; lanes 0-15 take K=[0..7]+[16..23], lanes 16-31 K=[8..15]+[24..31]
__device__ __forceinline__ v16bf load_fragA(const __bf16* s, int row0, int lane) {
  const int m  = lane & 15;
  const int ko = (lane >> 4) << 3;
  const __bf16* p = s + (row0 + m) * PITCH + ko;
  union U { v16bf v; v4f f[2]; } u;
  u.f[0] = *(const v4f*)(p);
  u.f[1] = *(const v4f*)(p + 16);
  return u.v;
}

// B fragment (32x16 bf16; column n is text row n, row-major in LDS):
// lane n=lane&15 holds col N=n; lanes 0-15 take K=[0..7]+[16..23], lanes 16-31 K=[8..15]+[24..31]
__device__ __forceinline__ v16bf load_fragB(const __bf16* s, int col0, int lane) {
  const int n  = lane & 15;
  const int kb = (lane >> 4) << 3;
  const __bf16* p = s + (col0 + n) * PITCH + kb;
  union U { v16bf v; v4f f[2]; } u;
  u.f[0] = *(const v4f*)(p);
  u.f[1] = *(const v4f*)(p + 16);
  return u.v;
}

__global__ __launch_bounds__(256) void siglip_tile_kernel(
    const float* __restrict__ img, const float* __restrict__ txt,
    const float* __restrict__ lscale, const float* __restrict__ lbias,
    float* __restrict__ partial) {
  __shared__ __align__(16) __bf16 sm[SM_ELEMS];   // 72 KB

  const int tid  = threadIdx.x;
  const int lane = tid & 31;
  const int wave = tid >> 5;

  const int by = blockIdx.y;          // M tile index (256 rows)
  const int bx = blockIdx.x;          // N tile index (128 cols)

  // staging mapping: A -> one full 32-wide row per thread; B -> half row per thread
  const int ra  = tid;                // 0..255
  const int rb  = tid >> 1;           // 0..127
  const int cbb = (tid & 1) * 16;

  const float* aRow = img + (size_t)(by * BM + ra) * DDIM;
  const float* bRow = txt + (size_t)(bx * BN + rb) * DDIM + cbb;

  // register prefetch of k-slab 0
  v4f pa[8], pb[4];
#pragma unroll
  for (int i = 0; i < 8; ++i) pa[i] = *(const v4f*)(aRow + 4 * i);
#pragma unroll
  for (int i = 0; i < 4; ++i) pb[i] = *(const v4f*)(bRow + 4 * i);

  const int wm = wave & 3;            // 4 waves along M: 64 rows each
  const int wn = wave >> 2;           // 2 waves along N: 64 cols each
  const int wrow = wm * 64;
  const int wcol = wn * 64;

  v8f acc[4][4];
#pragma unroll
  for (int mt = 0; mt < 4; ++mt)
#pragma unroll
    for (int nt = 0; nt < 4; ++nt)
#pragma unroll
      for (int j = 0; j < 8; ++j) acc[mt][nt][j] = 0.0f;

  const int KSTEPS = DDIM / BK;       // 16
  for (int ks = 0; ks < KSTEPS; ++ks) {
    __syncthreads();                  // previous iteration done reading LDS
    // stage prefetched regs -> LDS with split-bf16 (hi + lo) decomposition
    {
      union SA { __bf16 h[32]; v4f f[4]; } ah, al;
      union SB { __bf16 h[16]; v4f f[2]; } bh, bl;
#pragma unroll
      for (int i = 0; i < 8; ++i) {
        v4f q = pa[i];
#pragma unroll
        for (int c = 0; c < 4; ++c) {
          float x = q[c];
          __bf16 h = (__bf16)x;
          ah.h[4 * i + c] = h;
          al.h[4 * i + c] = (__bf16)(x - (float)h);
        }
      }
#pragma unroll
      for (int i = 0; i < 4; ++i) {
        v4f q = pb[i];
#pragma unroll
        for (int c = 0; c < 4; ++c) {
          float x = q[c];
          __bf16 h = (__bf16)x;
          bh.h[4 * i + c] = h;
          bl.h[4 * i + c] = (__bf16)(x - (float)h);
        }
      }
      __bf16* pAhi = &sm[AHI + ra * PITCH];
      __bf16* pAlo = &sm[ALO + ra * PITCH];
#pragma unroll
      for (int i = 0; i < 4; ++i) {
        *(v4f*)(pAhi + 8 * i) = ah.f[i];
        *(v4f*)(pAlo + 8 * i) = al.f[i];
      }
      __bf16* pBhi = &sm[BHI + rb * PITCH + cbb];
      __bf16* pBlo = &sm[BLO + rb * PITCH + cbb];
      *(v4f*)(pBhi) = bh.f[0]; *(v4f*)(pBhi + 8) = bh.f[1];
      *(v4f*)(pBlo) = bl.f[0]; *(v4f*)(pBlo + 8) = bl.f[1];
    }
    __syncthreads();                  // tile visible to all waves

    // issue next k-slab's global loads; latency hidden behind the WMMAs below
    if (ks + 1 < KSTEPS) {
      const int ko = (ks + 1) * BK;
#pragma unroll
      for (int i = 0; i < 8; ++i) pa[i] = *(const v4f*)(aRow + ko + 4 * i);
#pragma unroll
      for (int i = 0; i < 4; ++i) pb[i] = *(const v4f*)(bRow + ko + 4 * i);
    }

    v16bf ahf[4], alf[4];
#pragma unroll
    for (int mt = 0; mt < 4; ++mt) {
      ahf[mt] = load_fragA(&sm[AHI], wrow + mt * 16, lane);
      alf[mt] = load_fragA(&sm[ALO], wrow + mt * 16, lane);
    }
#pragma unroll
    for (int nt = 0; nt < 4; ++nt) {
      v16bf bhf = load_fragB(&sm[BHI], wcol + nt * 16, lane);
      v16bf blf = load_fragB(&sm[BLO], wcol + nt * 16, lane);
#pragma unroll
      for (int mt = 0; mt < 4; ++mt) {
        // hi*hi + lo*hi + hi*lo  (lo*lo dropped) ~= fp32 product
        acc[mt][nt] = __builtin_amdgcn_wmma_f32_16x16x32_bf16(
            false, ahf[mt], false, bhf, (short)0, acc[mt][nt], false, false);
        acc[mt][nt] = __builtin_amdgcn_wmma_f32_16x16x32_bf16(
            false, alf[mt], false, bhf, (short)0, acc[mt][nt], false, false);
        acc[mt][nt] = __builtin_amdgcn_wmma_f32_16x16x32_bf16(
            false, ahf[mt], false, blf, (short)0, acc[mt][nt], false, false);
      }
    }
  }

  // epilogue: logits -> stable log-sigmoid, accumulate per-thread partial
  const float escale = __expf(lscale[0]);
  const float bias   = lbias[0];
  const int rsel = (lane >> 4) * 8;   // C/D layout: lanes 16-31 hold rows M+8
  const int csel = lane & 15;
  float p = 0.0f;
#pragma unroll
  for (int mt = 0; mt < 4; ++mt) {
#pragma unroll
    for (int nt = 0; nt < 4; ++nt) {
#pragma unroll
      for (int j = 0; j < 8; ++j) {
        int row = by * BM + wrow + mt * 16 + j + rsel;
        int col = bx * BN + wcol + nt * 16 + csel;
        float v = acc[mt][nt][j] * escale + bias;
        float x = (row == col) ? v : -v;                    // labels = 2I - 1
        p += fminf(x, 0.0f) - __logf(1.0f + __expf(-fabsf(x)));
      }
    }
  }
  // wave32 shuffle reduce
#pragma unroll
  for (int off = 16; off > 0; off >>= 1) p += __shfl_down(p, off, 32);

  __syncthreads();
  float* red = (float*)sm;            // reuse LDS (all tile reads complete)
  if (lane == 0) red[wave] = p;
  __syncthreads();
  if (tid == 0) {
    float s = 0.0f;
#pragma unroll
    for (int w = 0; w < 8; ++w) s += red[w];
    partial[by * gridDim.x + bx] = s; // deterministic per-block slot
  }
}

__global__ void siglip_finalize_kernel(const float* __restrict__ partial,
                                       float* __restrict__ out, int nparts) {
  int lane = threadIdx.x;             // single wave of 32
  float s = 0.0f;
  for (int i = lane; i < nparts; i += 32) s += partial[i];
#pragma unroll
  for (int off = 16; off > 0; off >>= 1) s += __shfl_down(s, off, 32);
  if (lane == 0) out[0] = -s / (float)NTOT;  // (loss_img + loss_txt)/2 == loss_img
}

extern "C" void kernel_launch(void* const* d_in, const int* in_sizes, int n_in,
                              void* d_out, int out_size, void* d_ws, size_t ws_size,
                              hipStream_t stream) {
  const float* img    = (const float*)d_in[0];
  const float* txt    = (const float*)d_in[1];
  const float* lscale = (const float*)d_in[2];
  const float* lbias  = (const float*)d_in[3];
  float* out     = (float*)d_out;
  float* partial = (float*)d_ws;      // 128*64 floats = 32 KB

  dim3 grid(NTOT / BN, NTOT / BM);    // (128, 64)
  dim3 block(256);
  siglip_tile_kernel<<<grid, block, 0, stream>>>(img, txt, lscale, lbias, partial);
  siglip_finalize_kernel<<<1, 32, 0, stream>>>(partial, out,
                                               (NTOT / BM) * (NTOT / BN));
}
